// d2nn_layer_90013924589952
// MI455X (gfx1250) — matrix operand order repaired and checked by reading, not compiled, not run on hardware
//
#include <hip/hip_runtime.h>
#include <stdint.h>
#include <stddef.h>

// Problem constants (B=32, NI=64, NO=64)
#define NPIX   4096          // K = NI*NI
#define NOUTN  4096          // N = NO*NO
#define KCHUNK 32            // K per WMMA step
#define NCHUNKS 128          // NPIX / KCHUNK
#define KSEG   8             // K split for parallelism
#define CSEG   16            // chunks per segment (NCHUNKS/KSEG)
#define NTILE  32            // N columns per workgroup (one 128B line per row)
#define LSTRIDE 36           // padded LDS row stride (floats), 16B aligned
#define COMPF  (KCHUNK*LSTRIDE + 16)  // floats per component per buffer

typedef __attribute__((ext_vector_type(16))) _Float16 v16h;
typedef __attribute__((ext_vector_type(8)))  float    v8f;
typedef __attribute__((ext_vector_type(2)))  _Float16 h2;
typedef __attribute__((ext_vector_type(4)))  unsigned int u32x4;
typedef __attribute__((ext_vector_type(4)))  float    f32x4;
typedef __attribute__((ext_vector_type(4)))  int      i32x4;

#if __has_builtin(__builtin_amdgcn_global_load_async_to_lds_b128)
#define HAVE_ASYNC 1
typedef __attribute__((address_space(1))) i32x4 g_i32x4;
typedef __attribute__((address_space(3))) i32x4 l_i32x4;
#endif

union AFrag { v16h v; u32x4 q[2]; };
union BFrag { v16h v; unsigned int w[8]; };

// pack two f32 -> one dword of two f16 (v_cvt_pk_rtz_f16_f32)
__device__ __forceinline__ unsigned int cvt2u(float x, float y) {
#if __has_builtin(__builtin_amdgcn_cvt_pkrtz)
  return __builtin_bit_cast(unsigned int, __builtin_amdgcn_cvt_pkrtz(x, y));
#else
  h2 r; r.x = (_Float16)x; r.y = (_Float16)y;
  return __builtin_bit_cast(unsigned int, r);
#endif
}

template<int N>
__device__ __forceinline__ void wait_async() {
#ifdef HAVE_ASYNC
#if __has_builtin(__builtin_amdgcn_s_wait_asynccnt)
  __builtin_amdgcn_s_wait_asynccnt(N);
#else
  asm volatile("s_wait_asynccnt %0" :: "n"(N) : "memory");
#endif
#endif
}

// padded LDS row offset: row*36 + 16 extra floats for rows 16..31
// => lanes 0-15 and 16-31 of a wave hit disjoint bank groups on column reads
__device__ __forceinline__ int row_off(int r) { return r * LSTRIDE + ((r >> 4) << 4); }

// Copy one K-chunk W tile (32 K-rows x 32 N-cols, re+im, f32) global -> LDS.
// 64 threads, 8 async transfers of 16B each (512 quads total).
__device__ __forceinline__ void load_tile(const float* __restrict__ wre,
                                          const float* __restrict__ wim,
                                          int k0, int n0, float* lbuf, int tid) {
#pragma unroll
  for (int comp = 0; comp < 2; ++comp) {
    const float* W = comp ? wim : wre;
    float* L = lbuf + comp * COMPF;
#pragma unroll
    for (int i = 0; i < 4; ++i) {
      int flat = i * 64 + tid;
      int row = flat >> 3;          // 0..31
      int q   = flat & 7;           // 0..7 (16B quad within the 128B row)
      const float* g = W + (size_t)(k0 + row) * NOUTN + n0 + q * 4;
      float*       l = L + row_off(row) + q * 4;
#ifdef HAVE_ASYNC
      __builtin_amdgcn_global_load_async_to_lds_b128(
          (g_i32x4*)(uintptr_t)g, (l_i32x4*)(uint32_t)(uintptr_t)l, 0, 0);
#else
      *(f32x4*)l = *(const f32x4*)g;
#endif
    }
  }
}

// ---------------------------------------------------------------------------
// Kernel 1: modulated field -> f16 WMMA A-fragments in d_ws.
// A-fragment layout (ISA 16-bit A 16x32 table): for K kk in chunk,
// g=kk/8, o=kk%8: lane = row + 16*(g&1), half = o + 8*(g>>1).
// ws block (comp,tile,chunk): 32 lanes x 16 halves = 1KB; lane loads 32B flat.
// ---------------------------------------------------------------------------
__global__ __launch_bounds__(256) void d2nn_modulate(
    const float* __restrict__ ire, const float* __restrict__ iim,
    const float* __restrict__ amp, const float* __restrict__ ph,
    _Float16* __restrict__ afrag) {
  int idx = blockIdx.x * 256 + threadIdx.x;      // 32*4096 threads exactly
  int b = idx >> 12;
  int k = idx & (NPIX - 1);
  float a = amp[k], p = ph[k];
  float sg = 1.0f / (1.0f + __expf(-a));
  float sn, cs; __sincosf(p, &sn, &cs);
  float tr = sg * cs, ti = sg * sn;
  float er = ire[idx], ei = iim[idx];
  float mr = er * tr - ei * ti;
  float mi = er * ti + ei * tr;
  int t = b >> 4, r = b & 15;
  int c = k >> 5, kk = k & 31;
  int g = kk >> 3, o = kk & 7;
  int lane = r + ((g & 1) << 4);
  int h = o + ((g >> 1) << 3);
  size_t base = ((size_t)t * NCHUNKS + c) * 512 + (size_t)lane * 16 + h;
  afrag[base] = (_Float16)mr;                               // re component
  afrag[(size_t)2 * NCHUNKS * 512 + base] = (_Float16)mi;   // im component
}

// ---------------------------------------------------------------------------
// Kernel 2: complex GEMM via v_wmma_f32_16x16x32_f16.
// grid (128 N-tiles, 8 K-segments); block 64 = 2 waves.
// wave u handles 16 N columns and BOTH M-tiles (B fragments built once,
// reused by 8 WMMAs per chunk).
// Pt += Are_t*Wre; Qt += Aim_t*Wim; Rt += Are_t*Wim + Aim_t*Wre;
// out_re = P-Q, out_im = R.
// ---------------------------------------------------------------------------
__global__ __launch_bounds__(64) void d2nn_gemm(
    const float* __restrict__ wre, const float* __restrict__ wim,
    const _Float16* __restrict__ afrag, float* __restrict__ out) {
  __shared__ __align__(16) float lds[2][2][COMPF];   // [buf][comp][padded tile]
  const int tid  = threadIdx.x;
  const int lane = tid & 31;
  const int u    = tid >> 5;          // wave id = N sub-tile (16 cols)
  const int n0 = blockIdx.x * NTILE;
  const int c0 = blockIdx.y * CSEG;

  v8f P0 = {}; v8f Q0 = {}; v8f R0 = {};
  v8f P1 = {}; v8f Q1 = {}; v8f R1 = {};

  // prologue: fill buffer 0
  load_tile(wre, wim, c0 * KCHUNK, n0, &lds[0][0][0], tid);

  for (int cl = 0; cl < CSEG; ++cl) {
    const int gc = c0 + cl;
    if (cl + 1 < CSEG) {
      load_tile(wre, wim, (gc + 1) * KCHUNK, n0, &lds[(cl + 1) & 1][0][0], tid);
      wait_async<8>();   // 8 in flight for next tile; current tile complete
    } else {
      wait_async<0>();
    }
    __syncthreads();

    // ---- build B fragments from LDS (f32 -> packed f16), built ONCE ----
    const float* Lre = &lds[cl & 1][0][0];
    const float* Lim = &lds[cl & 1][1][0];
    const int nn = u * 16 + (lane & 15);     // column within 32-wide tile
    const int kh = (lane >> 4) << 3;         // K offset: 0 or 8 (table order)
    BFrag bre, bim;
#pragma unroll
    for (int s = 0; s < 8; ++s) {
      int o0 = row_off(kh + 2 * s + ((s >> 2) << 3));
      int o1 = o0 + LSTRIDE;
      bre.w[s] = cvt2u(Lre[o0 + nn], Lre[o1 + nn]);
      bim.w[s] = cvt2u(Lim[o0 + nn], Lim[o1 + nn]);
    }

    // ---- load A fragments for both M-tiles (pre-swizzled f16, L2-hot) ----
    AFrag are0, aim0, are1, aim1;
    {
      const _Float16* pr0 = afrag + ((size_t)0 * NCHUNKS + gc) * 512 + (size_t)lane * 16;
      const _Float16* pr1 = afrag + ((size_t)1 * NCHUNKS + gc) * 512 + (size_t)lane * 16;
      const u32x4* a0 = (const u32x4*)pr0;
      const u32x4* a1 = (const u32x4*)pr1;
      const u32x4* b0 = (const u32x4*)(pr0 + (size_t)2 * NCHUNKS * 512);
      const u32x4* b1 = (const u32x4*)(pr1 + (size_t)2 * NCHUNKS * 512);
      are0.q[0] = a0[0]; are0.q[1] = a0[1];
      are1.q[0] = a1[0]; are1.q[1] = a1[1];
      aim0.q[0] = b0[0]; aim0.q[1] = b0[1];
      aim1.q[0] = b1[0]; aim1.q[1] = b1[1];
    }

    // ---- 8 WMMAs per chunk (B fragments reused 4x each) ----
    P0 = __builtin_amdgcn_wmma_f32_16x16x32_f16(false, are0.v, false, bre.v, (short)0, P0, false, false);
    Q0 = __builtin_amdgcn_wmma_f32_16x16x32_f16(false, aim0.v, false, bim.v, (short)0, Q0, false, false);
    R0 = __builtin_amdgcn_wmma_f32_16x16x32_f16(false, are0.v, false, bim.v, (short)0, R0, false, false);
    R0 = __builtin_amdgcn_wmma_f32_16x16x32_f16(false, aim0.v, false, bre.v, (short)0, R0, false, false);
    P1 = __builtin_amdgcn_wmma_f32_16x16x32_f16(false, are1.v, false, bre.v, (short)0, P1, false, false);
    Q1 = __builtin_amdgcn_wmma_f32_16x16x32_f16(false, aim1.v, false, bim.v, (short)0, Q1, false, false);
    R1 = __builtin_amdgcn_wmma_f32_16x16x32_f16(false, are1.v, false, bim.v, (short)0, R1, false, false);
    R1 = __builtin_amdgcn_wmma_f32_16x16x32_f16(false, aim1.v, false, bre.v, (short)0, R1, false, false);

    __syncthreads();   // protect current buffer before next iteration refills it
  }

  // ---- epilogue: C layout lanes0-15: M=v, lanes16-31: M=8+v; N=lane&15 ----
  const int mbase = (lane >> 4) << 3;
  const int no = n0 + u * 16 + (lane & 15);
#pragma unroll
  for (int v = 0; v < 8; ++v) {
    size_t o0 = ((size_t)(mbase + v) * NOUTN + no) * 2;          // M-tile 0
    size_t o1 = ((size_t)(16 + mbase + v) * NOUTN + no) * 2;     // M-tile 1
    unsafeAtomicAdd(&out[o0],     P0[v] - Q0[v]);
    unsafeAtomicAdd(&out[o0 + 1], R0[v]);
    unsafeAtomicAdd(&out[o1],     P1[v] - Q1[v]);
    unsafeAtomicAdd(&out[o1 + 1], R1[v]);
  }
}

// ---------------------------------------------------------------------------
extern "C" void kernel_launch(void* const* d_in, const int* in_sizes, int n_in,
                              void* d_out, int out_size, void* d_ws, size_t ws_size,
                              hipStream_t stream) {
  const float* ire = (const float*)d_in[0];
  const float* iim = (const float*)d_in[1];
  const float* wre = (const float*)d_in[2];
  const float* wim = (const float*)d_in[3];
  const float* amp = (const float*)d_in[4];
  const float* ph  = (const float*)d_in[5];
  float* out = (float*)d_out;
  _Float16* afrag = (_Float16*)d_ws;   // needs 512 KB of scratch

  // zero accumulation target (32*4096 complex64 = 1 MB)
  (void)hipMemsetAsync(d_out, 0, (size_t)32 * NOUTN * 2 * sizeof(float), stream);

  // 32*4096 elements, one thread each
  d2nn_modulate<<<dim3((32 * NPIX) / 256), dim3(256), 0, stream>>>(ire, iim, amp, ph, afrag);

  // 128 N-tiles x 8 K-segments, 64 threads (2 waves)
  d2nn_gemm<<<dim3(NOUTN / NTILE, KSEG), dim3(64), 0, stream>>>(wre, wim, afrag, out);
}
